// NewFormer_12094627905737
// MI455X (gfx1250) — compile-verified, matrix operand order checked by reading, not run on hardware
//
#include <hip/hip_runtime.h>
#include <math.h>

// ---------------------------------------------------------------------------
// CDNA5 (gfx1250) wave32 WMMA / TDM types
// ---------------------------------------------------------------------------
typedef __attribute__((ext_vector_type(16))) _Float16 v16h;
typedef __attribute__((ext_vector_type(8)))  float    v8f;
typedef __attribute__((ext_vector_type(4)))  unsigned int u32x4;
typedef __attribute__((ext_vector_type(8)))  int      i32x8;
typedef __attribute__((ext_vector_type(4)))  int      i32x4;

// ---------------------------------------------------------------------------
// Fragment load helpers. ISA 16-bit operand layout (16x32 tile, wave32):
//   lane row/col = lane&15 ; element e -> K offset:
//     e<8 : 8*half + e          (contiguous 8 floats)
//     e>=8: 16 + 8*half + (e-8) (contiguous 8 floats)
// Direct path: per-lane contiguous -> 4x global_load_b128.
// ---------------------------------------------------------------------------
__device__ __forceinline__ v16h frag_direct(const float* __restrict__ base, int half)
{
    const float4* p0 = (const float4*)(base + 8 * half);
    const float4* p1 = (const float4*)(base + 16 + 8 * half);
    float4 r0 = p0[0], r1 = p0[1], r2 = p1[0], r3 = p1[1];
    v16h f;
    f[0]  = (_Float16)r0.x; f[1]  = (_Float16)r0.y; f[2]  = (_Float16)r0.z; f[3]  = (_Float16)r0.w;
    f[4]  = (_Float16)r1.x; f[5]  = (_Float16)r1.y; f[6]  = (_Float16)r1.z; f[7]  = (_Float16)r1.w;
    f[8]  = (_Float16)r2.x; f[9]  = (_Float16)r2.y; f[10] = (_Float16)r2.z; f[11] = (_Float16)r2.w;
    f[12] = (_Float16)r3.x; f[13] = (_Float16)r3.y; f[14] = (_Float16)r3.z; f[15] = (_Float16)r3.w;
    return f;
}

__device__ __forceinline__ v16h frag_from_tile(const float (*tile)[17], int half, int lr)
{
    v16h f;
#pragma unroll
    for (int e = 0; e < 16; ++e) {
        int k = (e < 8) ? (8 * half + e) : (8 + 8 * half + e);
        f[e] = (_Float16)tile[k][lr];
    }
    return f;
}

// ---------------------------------------------------------------------------
// Tensor Data Mover: DMA one 32x16 f32 K-tile (row stride ld elements) from
// global into LDS, padded 1 DWORD per 16-DWORD row -> LDS tile[32][17].
// D# per CDNA5 ISA 8.3/8.4: group0 {count=1, lds_addr, global_addr, type=2},
// group1 {data_size=4B, pad_en, pad_interval=16dw, pad_amount=1dw,
//         tensor_dim0=16, tensor_dim1=32, tile_dim0=16, tile_dim1=32,
//         tensor_dim0_stride=ld}. 2-D tensor -> groups 2/3 (and trailing
// group) zero. Completion tracked with TENSORcnt.
// This toolchain exposes the 6-arg builtin:
//   (u32x4 g0, i32x8 g1, i32x4 g2, i32x4 g3, i32x8 g4, i32 cpol)
// ---------------------------------------------------------------------------
__device__ __forceinline__ void tdm_load_tile(const float* src, int ld, void* ldsDst)
{
    unsigned long long ga = (unsigned long long)(uintptr_t)src;
    unsigned int lds = (unsigned int)(uintptr_t)ldsDst;  // LDS aperture: low 32 bits
    u32x4 g0;
    g0[0] = 1u;                                              // count=1 (valid user D#)
    g0[1] = lds;                                             // lds_addr (bytes)
    g0[2] = (unsigned int)ga;                                // global_addr[31:0]
    g0[3] = (unsigned int)((ga >> 32) & 0x1FFFFFFu) | (2u << 30);  // [56:32] | type=2
    i32x8 g1;
    g1[0] = (int)((2u << 16) | (1u << 20) | (3u << 22));     // 4B | pad_en | 16dw interval
    g1[1] = (int)(16u << 16);                                // tensor_dim0 = 16
    g1[2] = (int)(32u << 16);                                // tensor_dim1 = 32
    g1[3] = (int)(16u << 16);                                // tile_dim0 = 16
    g1[4] = 32;                                              // tile_dim1 = 32
    g1[5] = (int)(unsigned int)ld;                           // dim0 stride (elems) lo32
    g1[6] = 0;
    g1[7] = 0;
    i32x4 g2 = {0, 0, 0, 0}, g3 = {0, 0, 0, 0};
    i32x8 g4 = {0, 0, 0, 0, 0, 0, 0, 0};
    __builtin_amdgcn_tensor_load_to_lds(g0, g1, g2, g3, g4, 0);
}

// ---------------------------------------------------------------------------
// Generic strided/batched WMMA GEMM: C[z][M,N] = A[z](MxK) * B[z](KxN) + epi
//   TRANSA: A stored (K x M): A[k*lda+m] -> TDM-staged through LDS
//   TRANSB: B stored (N x K): B[n*ldb+k] -> direct; else TDM-staged
//   EPI: 0 none | 1 +bias | 2 +bias,GELU(exact) | 3 +bias,+res | 4 *scale
// Requires K % 32 == 0, M,N % 16 == 0. One wave per 16x16 tile -> EXEC all-1s.
// Staged tiles are double-buffered: TDM load of tile i+1 is issued before the
// s_wait_tensorcnt on tile i, overlapping DMA with WMMA + direct VMEM loads.
// ---------------------------------------------------------------------------
template<int TRANSA, int TRANSB, int EPI>
__global__ __launch_bounds__(32) void gemm_wmma_k(
    const float* __restrict__ A, int lda, size_t aZ,
    const float* __restrict__ B, int ldb, size_t bZ,
    const float* __restrict__ bias, const float* __restrict__ res,
    float* __restrict__ C, int ldc, size_t cZ,
    int M, int N, int K, float scale)
{
    constexpr int STAGE_A = TRANSA;
    constexpr int STAGE_B = !TRANSB;
    __shared__ float tA[2][32][17];
    __shared__ float tB[2][32][17];
    const int lane = threadIdx.x & 31;
    const int half = lane >> 4;
    const int lr   = lane & 15;
    const int m0 = blockIdx.y << 4;
    const int n0 = blockIdx.x << 4;
    const size_t z = blockIdx.z;
    A += z * aZ; B += z * bZ; C += z * cZ;

    const float* aRow = A + (size_t)(m0 + lr) * lda;   // used if !TRANSA
    const float* bRow = B + (size_t)(n0 + lr) * ldb;   // used if  TRANSB

    if (STAGE_A) tdm_load_tile(A + m0, lda, tA[0]);
    if (STAGE_B) tdm_load_tile(B + n0, ldb, tB[0]);

    int buf = 0;
    v8f acc = {};
    for (int k0 = 0; k0 < K; k0 += 32) {
        const int nk = k0 + 32;
        if (nk < K) {   // issue next staged tile(s) before waiting on current
            if (STAGE_A) tdm_load_tile(A + (size_t)nk * lda + m0, lda, tA[buf ^ 1]);
            if (STAGE_B) tdm_load_tile(B + (size_t)nk * ldb + n0, ldb, tB[buf ^ 1]);
            __builtin_prefetch(TRANSB ? (const void*)(bRow + nk)
                                      : (const void*)(aRow + nk), 0, 1);
        }
        if (STAGE_A || STAGE_B) {
            if (nk < K) __builtin_amdgcn_s_wait_tensorcnt(STAGE_A + STAGE_B);
            else        __builtin_amdgcn_s_wait_tensorcnt(0);
            asm volatile("" ::: "memory");   // keep LDS reads behind the wait
        }
        v16h a = TRANSA ? frag_from_tile(tA[buf], half, lr) : frag_direct(aRow + k0, half);
        v16h b = TRANSB ? frag_direct(bRow + k0, half) : frag_from_tile(tB[buf], half, lr);
        acc = __builtin_amdgcn_wmma_f32_16x16x32_f16(
            /*neg_a=*/false, a, /*neg_b=*/false, b,
            /*c_mod=*/(short)0, acc, /*reuse_a=*/false, /*reuse_b=*/false);
        buf ^= 1;
    }

    const int n = n0 + lr;
#pragma unroll
    for (int r = 0; r < 8; ++r) {
        const int m = m0 + r + 8 * half;
        float v = acc[r];
        if (EPI == 1 || EPI == 2 || EPI == 3) v += bias[n];
        if (EPI == 2) v = 0.5f * v * (1.0f + erff(v * 0.70710678118654752f));
        if (EPI == 3) v += res[(size_t)m * ldc + n];
        if (EPI == 4) v *= scale;
        C[(size_t)m * ldc + n] = v;
    }
}

// ---------------------------------------------------------------------------
// LayerNorm over D elems of each row (row stride ld), eps=1e-5, optional add.
// One wave per row.
// ---------------------------------------------------------------------------
__global__ __launch_bounds__(32) void layernorm_k(
    const float* __restrict__ x, int ld, const float* __restrict__ g,
    const float* __restrict__ b, const float* __restrict__ add,
    float* __restrict__ y, int N, int D)
{
    const int row = blockIdx.x;
    if (row >= N) return;
    const float* xr = x + (size_t)row * ld;
    float s = 0.0f, s2 = 0.0f;
    for (int i = threadIdx.x; i < D; i += 32) { float v = xr[i]; s += v; s2 += v * v; }
#pragma unroll
    for (int off = 16; off; off >>= 1) {
        s  += __shfl_xor(s,  off, 32);
        s2 += __shfl_xor(s2, off, 32);
    }
    const float mean = s / (float)D;
    const float var  = s2 / (float)D - mean * mean;
    const float rstd = rsqrtf(var + 1e-5f);
    float* yr = y + (size_t)row * ld;
    for (int i = threadIdx.x; i < D; i += 32) {
        float v = (xr[i] - mean) * rstd * g[i] + b[i];
        if (add) v += add[(size_t)row * ld + i];
        yr[i] = v;
    }
}

// Row-wise softmax in place; gridDim.x = rows, one wave per row.
__global__ __launch_bounds__(32) void softmax_k(float* __restrict__ S, int N)
{
    float* row = S + (size_t)blockIdx.x * N;
    float lmax = -3.4e38f;
    for (int j = threadIdx.x; j < N; j += 32) lmax = fmaxf(lmax, row[j]);
#pragma unroll
    for (int off = 16; off; off >>= 1) lmax = fmaxf(lmax, __shfl_xor(lmax, off, 32));
    float lsum = 0.0f;
    for (int j = threadIdx.x; j < N; j += 32) { float p = expf(row[j] - lmax); row[j] = p; lsum += p; }
#pragma unroll
    for (int off = 16; off; off >>= 1) lsum += __shfl_xor(lsum, off, 32);
    const float inv = 1.0f / lsum;
    for (int j = threadIdx.x; j < N; j += 32) row[j] *= inv;
}

// ---------------------------------------------------------------------------
// Patch gather into K=64 zero-padded rows: x (3,256,256) -> t0 (4096 x 64)
// ---------------------------------------------------------------------------
__global__ void patch_gather_k(const float* __restrict__ x, float* __restrict__ t0)
{
    int t = blockIdx.x * blockDim.x + threadIdx.x;
    if (t >= 4096 * 64) return;
    int tok = t >> 6, j = t & 63;
    float v = 0.0f;
    if (j < 48) {
        int h = tok >> 6, w = tok & 63;
        int p1 = j / 12, rem = j % 12, p2 = rem / 3, c = rem % 3;
        v = x[(size_t)c * 65536 + (size_t)(h * 4 + p1) * 256 + (w * 4 + p2)];
    }
    t0[t] = v;
}

// Zero-pad patch weight (48x96) -> (64x96)
__global__ void pad_w_k(const float* __restrict__ w, float* __restrict__ wp)
{
    int t = blockIdx.x * blockDim.x + threadIdx.x;
    if (t >= 64 * 96) return;
    int k = t / 96, n = t % 96;
    wp[t] = (k < 48) ? w[k * 96 + n] : 0.0f;
}

// unfold input gather: F (S_src x C token-major) -> P (W2*W2 x 4C), cp=c*4+k1*2+k2
__global__ void unfold_gather_k(const float* __restrict__ F, float* __restrict__ P,
                                int C, int W2)
{
    int total = W2 * W2 * C * 4;
    int t = blockIdx.x * blockDim.x + threadIdx.x;
    if (t >= total) return;
    int C4 = C * 4;
    int s2 = t / C4, cp = t % C4;
    int c = cp >> 2, k = cp & 3, k1 = k >> 1, k2 = k & 1;
    int h2 = s2 / W2, w2 = s2 % W2, W = W2 * 2;
    P[t] = F[(size_t)((2 * h2 + k1) * W + (2 * w2 + k2)) * C + c];
}

// Token selection (amap is provably uniform 1/N -> stable argsort = identity,
// so top-75% = prefix). src==nullptr means orig_idx = arange. row = IMG/tgt_ps.
__global__ void gen_idx_k(const int* __restrict__ src, int* __restrict__ dst,
                          int T, int row)
{
    int t = blockIdx.x * blockDim.x + threadIdx.x;
    if (t >= T) return;
    int i = t >> 2, q = t & 3;
    int top = src ? src[i] : i;
    int rw = row >> 1;
    int r = top / rw, c = top - r * rw;
    int i1 = 2 * r * row + 2 * c;
    int a = (q == 0) ? 0 : (q == 1) ? 1 : (q == 2) ? row : (row + 1);
    dst[t] = i1 + a;
}

__global__ void gather_k(const float* __restrict__ src, const int* __restrict__ idx,
                         float* __restrict__ dst, int T, int C)
{
    int t = blockIdx.x * blockDim.x + threadIdx.x;
    if (t >= T * C) return;
    int r = t / C, c = t % C;
    dst[t] = src[(size_t)idx[r] * C + c];
}

__global__ void scatter_k(float* __restrict__ dst, const int* __restrict__ idx,
                          const float* __restrict__ val, int T, int C)
{
    int t = blockIdx.x * blockDim.x + threadIdx.x;
    if (t >= T * C) return;
    int r = t / C, c = t % C;
    dst[(size_t)idx[r] * C + c] = val[t];
}

// nearest 2x upsample, token-major: src (Wc*Wc x C) -> dst (4*Wc*Wc x C)
__global__ void upsample2x_k(const float* __restrict__ src, float* __restrict__ dst,
                             int Wc, int C)
{
    int Wf = 2 * Wc;
    int total = Wf * Wf * C;
    int t = blockIdx.x * blockDim.x + threadIdx.x;
    if (t >= total) return;
    int s2 = t / C, c = t % C;
    int h2 = s2 / Wf, w2 = s2 % Wf;
    dst[t] = src[(size_t)((h2 >> 1) * Wc + (w2 >> 1)) * C + c];
}

// ---------------------------------------------------------------------------
// Host-side orchestration
// ---------------------------------------------------------------------------
struct LayerP {
    const float *ln_g, *ln_b, *w_qkv, *w_out, *b_out;
    const float *f_ln_g, *f_ln_b, *w1, *b1, *w2, *b2;
};

static void build_layers(void* const* d_in, int base, LayerP* L,
                         const float** n_g, const float** n_b)
{
    // sorted leaf order: b1,b2,b_out,f_ln_b,f_ln_g,ln_b,ln_g,w1,w2,w_out,w_qkv
    for (int l = 0; l < 4; ++l) {
        int b = base + l * 11;
        L[l].b1     = (const float*)d_in[b + 0];
        L[l].b2     = (const float*)d_in[b + 1];
        L[l].b_out  = (const float*)d_in[b + 2];
        L[l].f_ln_b = (const float*)d_in[b + 3];
        L[l].f_ln_g = (const float*)d_in[b + 4];
        L[l].ln_b   = (const float*)d_in[b + 5];
        L[l].ln_g   = (const float*)d_in[b + 6];
        L[l].w1     = (const float*)d_in[b + 7];
        L[l].w2     = (const float*)d_in[b + 8];
        L[l].w_out  = (const float*)d_in[b + 9];
        L[l].w_qkv  = (const float*)d_in[b + 10];
    }
    *n_b = (const float*)d_in[base + 44];
    *n_g = (const float*)d_in[base + 45];
}

static void run_transformer(float* x, float* fin, int N, int D, int heads, int dh,
                            const LayerP* L, const float* n_g, const float* n_b,
                            float* xn, float* qkvb, float* attb, float* scores,
                            float* ffnb, hipStream_t s)
{
    const int inner = heads * dh;
    const int tI = 3 * inner;
    const float scale = 1.0f / sqrtf((float)dh);
    for (int l = 0; l < 4; ++l) {
        const LayerP& p = L[l];
        layernorm_k<<<N, 32, 0, s>>>(x, D, p.ln_g, p.ln_b, nullptr, xn, N, D);
        // qkv = xn @ w_qkv                               (N x 3I)
        gemm_wmma_k<0, 0, 0><<<dim3(tI / 16, N / 16, 1), 32, 0, s>>>(
            xn, D, 0, p.w_qkv, tI, 0, nullptr, nullptr, qkvb, tI, 0, N, tI, D, 0.f);
        // scores[h] = scale * Q K^T : A=Q direct, B=K transposed-B (both contiguous)
        gemm_wmma_k<0, 1, 4><<<dim3(N / 16, N / 16, heads), 32, 0, s>>>(
            qkvb, tI, (size_t)dh, qkvb + inner, tI, (size_t)dh,
            nullptr, nullptr, scores, N, (size_t)N * N, N, N, dh, scale);
        softmax_k<<<heads * N, 32, 0, s>>>(scores, N);
        // attb[:, h*dh:(h+1)*dh] = P V : B=V normal form (TDM-staged)
        gemm_wmma_k<0, 0, 0><<<dim3(dh / 16, N / 16, heads), 32, 0, s>>>(
            scores, N, (size_t)N * N, qkvb + 2 * inner, tI, (size_t)dh,
            nullptr, nullptr, attb, inner, (size_t)dh, N, dh, N, 0.f);
        // x = attb @ w_out + b_out + x
        gemm_wmma_k<0, 0, 3><<<dim3(D / 16, N / 16, 1), 32, 0, s>>>(
            attb, inner, 0, p.w_out, D, 0, p.b_out, x, x, D, 0, N, D, inner, 0.f);
        layernorm_k<<<N, 32, 0, s>>>(x, D, p.f_ln_g, p.f_ln_b, nullptr, xn, N, D);
        // ffn = gelu(xn @ w1 + b1)
        gemm_wmma_k<0, 0, 2><<<dim3(4 * D / 16, N / 16, 1), 32, 0, s>>>(
            xn, D, 0, p.w1, 4 * D, 0, p.b1, nullptr, ffnb, 4 * D, 0, N, 4 * D, D, 0.f);
        // x = ffn @ w2 + b2 + x
        gemm_wmma_k<0, 0, 3><<<dim3(D / 16, N / 16, 1), 32, 0, s>>>(
            ffnb, 4 * D, 0, p.w2, D, 0, p.b2, x, x, D, 0, N, D, 4 * D, 0.f);
    }
    layernorm_k<<<N, 32, 0, s>>>(x, D, n_g, n_b, nullptr, fin, N, D);
}

static inline dim3 ew_grid(int n) { return dim3((n + 255) / 256); }

extern "C" void kernel_launch(void* const* d_in, const int* in_sizes, int n_in,
                              void* d_out, int out_size, void* d_ws, size_t ws_size,
                              hipStream_t stream)
{
    (void)in_sizes; (void)n_in; (void)out_size; (void)ws_size;
    // inputs (JAX pytree flatten: dicts by sorted key, tuples in order)
    const float* x_img   = (const float*)d_in[0];
    const float* patch_b = (const float*)d_in[1];
    const float* ln0_b   = (const float*)d_in[2];
    const float* ln0_g   = (const float*)d_in[3];
    const float* ln1_b   = (const float*)d_in[4];
    const float* ln1_g   = (const float*)d_in[5];
    const float* patch_w = (const float*)d_in[6];
    const float* pos     = (const float*)d_in[7];
    const float* r16_w   = (const float*)d_in[8];
    const float* r16_b   = (const float*)d_in[9];
    const float* r32_w   = (const float*)d_in[10];
    const float* r32_b   = (const float*)d_in[11];
    const float* r8_w    = (const float*)d_in[12];
    const float* r8_b    = (const float*)d_in[13];
    const float* u16_w   = (const float*)d_in[198];
    const float* u16_b   = (const float*)d_in[199];
    const float* u32_w   = (const float*)d_in[200];
    const float* u32_b   = (const float*)d_in[201];
    const float* u8_w    = (const float*)d_in[202];
    const float* u8_b    = (const float*)d_in[203];

    LayerP L32[4], L16[4], L8[4], L4[4];
    const float *ng32, *nb32, *ng16, *nb16, *ng8, *nb8, *ng4, *nb4;
    build_layers(d_in, 14,  L16, &ng16, &nb16);
    build_layers(d_in, 60,  L32, &ng32, &nb32);
    build_layers(d_in, 106, L4,  &ng4,  &nb4);
    build_layers(d_in, 152, L8,  &ng8,  &nb8);

    // workspace carve-up (floats; all chunk sizes multiples of 16 -> 16B-aligned)
    float* ws = (float*)d_ws;
    size_t o = 0;
    float* t0   = ws + o; o += 4096 * 64;        // patch tokens, K zero-padded
    float* pw   = ws + o; o += 64 * 96;          // padded patch weight
    float* tbuf = ws + o; o += 4096 * 96;        // t (4096 x 96)
    float* P    = ws + o; o += 4096 * 96;        // gather / embed scratch
    float* x16o = ws + o; o += 256 * 384;
    float* x8o  = ws + o; o += 1024 * 192;
    float* x32  = ws + o; o += 64 * 768;
    float* x16  = ws + o; o += 192 * 384;
    float* x8   = ws + o; o += 576 * 192;
    float* x4   = ws + o; o += 1728 * 96;
    float* xn   = ws + o; o += 1728 * 96;        // LN scratch (max N*D)
    float* qkvb = ws + o; o += 64 * 12288;       // max qkv (t32)
    float* attb = ws + o; o += 64 * 4096;        // max attn out (t32)
    float* ffnb = ws + o; o += 1728 * 384;       // max FFN hidden (t4)
    float* conv = ws + o; o += 1024 * 96;        // max coarse conv out
    float* scor = ws + o; o += 2 * 1728 * 1728;  // max heads*N*N (t4)
    int* idx16 = (int*)(ws + o); o += 192;
    int* idx8  = (int*)(ws + o); o += 576;
    int* idx4  = (int*)(ws + o); o += 1728;

    float* out0 = (float*)d_out;      // x32 final (64 x 768)
    float* out1 = out0 + 49152;       // x16_i     (256 x 384)
    float* out2 = out1 + 98304;       // x8_i      (1024 x 192)
    float* out3 = out2 + 196608;      // x4_i      (4096 x 96)

    hipStream_t s = stream;

    // patch embed: gather(+pad) -> LN(48) -> WMMA 64x96 GEMM (+b) -> LN(96)+pos
    patch_gather_k<<<ew_grid(4096 * 64), 256, 0, s>>>(x_img, t0);
    pad_w_k<<<ew_grid(64 * 96), 256, 0, s>>>(patch_w, pw);
    layernorm_k<<<4096, 32, 0, s>>>(t0, 64, ln0_g, ln0_b, nullptr, t0, 4096, 48);
    gemm_wmma_k<0, 0, 1><<<dim3(96 / 16, 4096 / 16, 1), 32, 0, s>>>(
        t0, 64, 0, pw, 96, 0, patch_b, nullptr, P, 96, 0, 4096, 96, 64, 0.f);
    layernorm_k<<<4096, 32, 0, s>>>(P, 96, ln1_g, ln1_b, pos, tbuf, 4096, 96);

    // unfold pyramid (gather -> WMMA GEMM with W(O,K) transposed-B, direct path)
    unfold_gather_k<<<ew_grid(1024 * 384), 256, 0, s>>>(tbuf, P, 96, 32);
    gemm_wmma_k<0, 1, 1><<<dim3(192 / 16, 1024 / 16, 1), 32, 0, s>>>(
        P, 384, 0, u8_w, 384, 0, u8_b, nullptr, x8o, 192, 0, 1024, 192, 384, 0.f);
    unfold_gather_k<<<ew_grid(256 * 768), 256, 0, s>>>(x8o, P, 192, 16);
    gemm_wmma_k<0, 1, 1><<<dim3(384 / 16, 256 / 16, 1), 32, 0, s>>>(
        P, 768, 0, u16_w, 768, 0, u16_b, nullptr, x16o, 384, 0, 256, 384, 768, 0.f);
    unfold_gather_k<<<ew_grid(64 * 1536), 256, 0, s>>>(x16o, P, 384, 8);
    gemm_wmma_k<0, 1, 1><<<dim3(768 / 16, 64 / 16, 1), 32, 0, s>>>(
        P, 1536, 0, u32_w, 1536, 0, u32_b, nullptr, x32, 768, 0, 64, 768, 1536, 0.f);

    // token indices (amap == 1/N uniform -> stable argsort = identity)
    gen_idx_k<<<1, 256, 0, s>>>(nullptr, idx16, 192, 16);
    gen_idx_k<<<3, 256, 0, s>>>(idx16, idx8, 576, 32);
    gen_idx_k<<<7, 256, 0, s>>>(idx8, idx4, 1728, 64);

    // transformers coarse -> fine; t32 final LN writes straight to d_out
    run_transformer(x32, out0, 64, 768, 16, 256, L32, ng32, nb32, xn, qkvb, attb, scor, ffnb, s);
    gather_k<<<ew_grid(192 * 384), 256, 0, s>>>(x16o, idx16, x16, 192, 384);
    run_transformer(x16, x16, 192, 384, 8, 128, L16, ng16, nb16, xn, qkvb, attb, scor, ffnb, s);
    gather_k<<<ew_grid(576 * 192), 256, 0, s>>>(x8o, idx8, x8, 576, 192);
    run_transformer(x8, x8, 576, 192, 4, 64, L8, ng8, nb8, xn, qkvb, attb, scor, ffnb, s);
    gather_k<<<ew_grid(1728 * 96), 256, 0, s>>>(tbuf, idx4, x4, 1728, 96);
    run_transformer(x4, x4, 1728, 96, 2, 32, L4, ng4, nb4, xn, qkvb, attb, scor, ffnb, s);

    // upsample cascade; raw (B,N,C)->(B,C,H,W) views == TRANSA GEMMs (TDM-staged)
    gemm_wmma_k<1, 1, 1><<<dim3(384 / 16, 64 / 16, 1), 32, 0, s>>>(
        out0, 64, 0, r32_w, 768, 0, r32_b, nullptr, conv, 384, 0, 64, 384, 768, 0.f);
    upsample2x_k<<<ew_grid(256 * 384), 256, 0, s>>>(conv, out1, 8, 384);
    scatter_k<<<ew_grid(192 * 384), 256, 0, s>>>(out1, idx16, x16, 192, 384);

    gemm_wmma_k<1, 1, 1><<<dim3(192 / 16, 256 / 16, 1), 32, 0, s>>>(
        out1, 256, 0, r16_w, 384, 0, r16_b, nullptr, conv, 192, 0, 256, 192, 384, 0.f);
    upsample2x_k<<<ew_grid(1024 * 192), 256, 0, s>>>(conv, out2, 16, 192);
    scatter_k<<<ew_grid(576 * 192), 256, 0, s>>>(out2, idx8, x8, 576, 192);

    gemm_wmma_k<1, 1, 1><<<dim3(96 / 16, 1024 / 16, 1), 32, 0, s>>>(
        out2, 1024, 0, r8_w, 192, 0, r8_b, nullptr, conv, 96, 0, 1024, 96, 192, 0.f);
    upsample2x_k<<<ew_grid(4096 * 96), 256, 0, s>>>(conv, out3, 32, 96);
    scatter_k<<<ew_grid(1728 * 96), 256, 0, s>>>(out3, idx4, x4, 1728, 96);
}
